// SelfAttentionHead_30966714204361
// MI455X (gfx1250) — matrix operand order repaired and checked by reading, not compiled, not run on hardware
//
#include <hip/hip_runtime.h>

// ---------------------------------------------------------------------------
// Fused causal self-attention head for MI455X (gfx1250, wave32, WMMA).
// One workgroup (512 threads = 16 waves) per batch element.
//   B=512, T=256, C=384, H=64.
// Single-pass projection: all three W^T staged in LDS (144 KB), x tile read
// once, each A load feeds 12 v_wmma_f32_16x16x32_bf16. Flash attention with
// online softmax on bf16 K/V^T held in LDS.
// ---------------------------------------------------------------------------

typedef __bf16 bf16_t;
typedef bf16_t v16bf __attribute__((ext_vector_type(16)));
typedef float  v8f   __attribute__((ext_vector_type(8)));

#define T_SZ 256
#define C_SZ 384
#define H_SZ 64
#define NKT  (C_SZ / 32)          // 12 k-steps of 32
#define WT_ELEMS (H_SZ * C_SZ)    // 24576 elems per staged W^T
#define NTHREADS 512

// A-matrix (16x32, 16-bit) lane-local K index for vector element e.
// Lanes 0-15: K 0..15; lanes 16-31: K 16..31 (contiguous, matches B reads).
__device__ __forceinline__ int a_col(int e, int half) {
  return 16 * half + e;
}

__device__ __forceinline__ v8f wmma_bf16(v16bf a, v16bf b, v8f c) {
  // (neg_a, A, neg_b, B, c_mod, C, reuse_a, reuse_b)
  return __builtin_amdgcn_wmma_f32_16x16x32_bf16(false, a, false, b,
                                                 (short)0, c, false, false);
}

// Stage W [C][H] (f32, global) into LDS as W^T [H][C] (bf16) so that WMMA
// B-operand reads are contiguous 32-byte ds loads.
__device__ __forceinline__ void fill_wt(bf16_t* s_wt,
                                        const float* __restrict__ W, int tid) {
  for (int i = tid; i < WT_ELEMS; i += NTHREADS) {
    int h = i / C_SZ;
    int c = i - h * C_SZ;
    s_wt[i] = (bf16_t)W[c * H_SZ + h];
  }
}

__global__ __launch_bounds__(NTHREADS, 1)
void fused_attn_head(const float* __restrict__ x,
                     const float* __restrict__ Wk,
                     const float* __restrict__ Wq,
                     const float* __restrict__ Wv,
                     float* __restrict__ out) {
  __shared__ __align__(64) bf16_t s_wt[3 * WT_ELEMS]; // 144 KB  Wq^T|Wk^T|Wv^T
  __shared__ __align__(64) bf16_t s_k [T_SZ * H_SZ];  //  32 KB  K  [T][H]
  __shared__ __align__(64) bf16_t s_vt[H_SZ * T_SZ];  //  32 KB  V^T[H][T]
  __shared__ __align__(64) bf16_t s_p [16 * 1024];    //  32 KB  per-wave scratch
                                                      //         (Q park 16x64, then P 16x32)

  const int tid  = threadIdx.x;
  const int lane = tid & 31;
  const int wave = tid >> 5;       // 0..15, owns rows [16*wave, 16*wave+16)
  const int half = lane >> 4;
  const int ln   = lane & 15;
  const int b    = blockIdx.x;
  const int m0   = wave * 16;
  const int pbase = wave * 1024;   // this wave's scratch region (elems)

  const float* xrow = x + ((size_t)b * T_SZ + (m0 + ln)) * C_SZ;

  // ---- Stage all three W^T (bf16) in LDS, one barrier. ----
  fill_wt(s_wt + 0 * WT_ELEMS, Wq, tid);
  fill_wt(s_wt + 1 * WT_ELEMS, Wk, tid);
  fill_wt(s_wt + 2 * WT_ELEMS, Wv, tid);
  __syncthreads();

  // ---- Fused Q/K/V projection: one pass over x, 12 WMMAs per A load. ----
  const v8f z8 = {0.f, 0.f, 0.f, 0.f, 0.f, 0.f, 0.f, 0.f};
  v8f aq[4], ak[4], av[4];
#pragma unroll
  for (int nt = 0; nt < 4; ++nt) { aq[nt] = z8; ak[nt] = z8; av[nt] = z8; }

#pragma unroll 1
  for (int kt = 0; kt < NKT; ++kt) {
    v16bf a;
#pragma unroll
    for (int e = 0; e < 16; ++e)
      a[e] = (bf16_t)xrow[kt * 32 + a_col(e, half)];
#pragma unroll
    for (int nt = 0; nt < 4; ++nt) {
      const int bo = (nt * 16 + ln) * C_SZ + kt * 32 + 16 * half;
      aq[nt] = wmma_bf16(a, *(const v16bf*)&s_wt[0 * WT_ELEMS + bo], aq[nt]);
      ak[nt] = wmma_bf16(a, *(const v16bf*)&s_wt[1 * WT_ELEMS + bo], ak[nt]);
      av[nt] = wmma_bf16(a, *(const v16bf*)&s_wt[2 * WT_ELEMS + bo], av[nt]);
    }
  }

  // ---- Scatter results: K -> s_k [T][H], V -> s_vt [H][T] (transposed). ----
#pragma unroll
  for (int nt = 0; nt < 4; ++nt)
#pragma unroll
    for (int r = 0; r < 8; ++r) {
      const int t = m0 + r + 8 * half;
      s_k [t * H_SZ + nt * 16 + ln]       = (bf16_t)ak[nt][r];
      s_vt[(nt * 16 + ln) * T_SZ + t]     = (bf16_t)av[nt][r];
      // Park Q (bf16) in this wave's private scratch, 16x64 row-major.
      s_p [pbase + (r + 8 * half) * H_SZ + nt * 16 + ln] = (bf16_t)aq[nt][r];
    }

  // Re-read Q in A layout (intra-wave round trip, only needs dscnt).
  asm volatile("s_wait_dscnt 0" ::: "memory");
  v16bf qa[2];
#pragma unroll
  for (int ks = 0; ks < 2; ++ks)
#pragma unroll
    for (int e = 0; e < 16; ++e)
      qa[ks][e] = s_p[pbase + ln * H_SZ + 32 * ks + a_col(e, half)];

  __syncthreads();  // K / V^T visible to all waves

  // ---------------- Flash attention over 32-wide column blocks -------------
  float mrow[8], lrow[8];
  v8f oacc[4];
#pragma unroll
  for (int nt = 0; nt < 4; ++nt) oacc[nt] = z8;
#pragma unroll
  for (int r = 0; r < 8; ++r) { mrow[r] = -__builtin_inff(); lrow[r] = 0.f; }

  for (int jp = 0; jp < T_SZ / 32; ++jp) {
    const int sbase = 32 * jp;
    if (sbase > m0 + 15) break;  // wave-uniform causal early exit

    float st[2][8];
#pragma unroll
    for (int jj = 0; jj < 2; ++jj) {
      const int s0 = sbase + 16 * jj;
      v8f sa = z8;
#pragma unroll
      for (int ks = 0; ks < 2; ++ks) {
        v16bf kb = *(const v16bf*)&s_k[(s0 + ln) * H_SZ + 32 * ks + 16 * half];
        sa = wmma_bf16(qa[ks], kb, sa);
      }
#pragma unroll
      for (int r = 0; r < 8; ++r) {
        const int qr = m0 + r + 8 * half;  // query row
        const int sc = s0 + ln;            // key column
        st[jj][r] = (sc <= qr) ? sa[r] * 0.125f : -__builtin_inff();
      }
    }

    // Online softmax update (rows live in 16-lane groups per half).
#pragma unroll
    for (int r = 0; r < 8; ++r) {
      float tm = fmaxf(st[0][r], st[1][r]);
      tm = fmaxf(tm, __shfl_xor(tm, 1));
      tm = fmaxf(tm, __shfl_xor(tm, 2));
      tm = fmaxf(tm, __shfl_xor(tm, 4));
      tm = fmaxf(tm, __shfl_xor(tm, 8));
      const float mnew = fmaxf(mrow[r], tm);
      const float corr = (mnew > -3.0e38f) ? __expf(mrow[r] - mnew) : 1.0f;
      const float p0 = (st[0][r] > -3.0e38f) ? __expf(st[0][r] - mnew) : 0.f;
      const float p1 = (st[1][r] > -3.0e38f) ? __expf(st[1][r] - mnew) : 0.f;
      float rs = p0 + p1;
      rs += __shfl_xor(rs, 1);
      rs += __shfl_xor(rs, 2);
      rs += __shfl_xor(rs, 4);
      rs += __shfl_xor(rs, 8);
      lrow[r] = lrow[r] * corr + rs;
      mrow[r] = mnew;
#pragma unroll
      for (int nt = 0; nt < 4; ++nt) oacc[nt][r] *= corr;
      s_p[pbase + (r + 8 * half) * 32 + ln]      = (bf16_t)p0;
      s_p[pbase + (r + 8 * half) * 32 + 16 + ln] = (bf16_t)p1;
    }

    // P (C-layout) -> A-layout via per-wave LDS scratch.
    asm volatile("s_wait_dscnt 0" ::: "memory");
    v16bf pa;
#pragma unroll
    for (int e = 0; e < 16; ++e)
      pa[e] = s_p[pbase + ln * 32 + a_col(e, half)];

    // O += P @ V  (K dim = 32 columns of this block)
#pragma unroll
    for (int nt = 0; nt < 4; ++nt) {
      v16bf vb = *(const v16bf*)&s_vt[(nt * 16 + ln) * T_SZ + sbase + 16 * half];
      oacc[nt] = wmma_bf16(pa, vb, oacc[nt]);
    }
  }

  // ---------------- Epilogue: normalize and store f32 ----------------
  float* ob = out + (size_t)b * T_SZ * H_SZ;
#pragma unroll
  for (int nt = 0; nt < 4; ++nt)
#pragma unroll
    for (int r = 0; r < 8; ++r) {
      const int t = m0 + r + 8 * half;
      ob[(size_t)t * H_SZ + nt * 16 + ln] = oacc[nt][r] / lrow[r];
    }
}

extern "C" void kernel_launch(void* const* d_in, const int* in_sizes, int n_in,
                              void* d_out, int out_size, void* d_ws, size_t ws_size,
                              hipStream_t stream) {
  const float* x  = (const float*)d_in[0];
  const float* Wk = (const float*)d_in[1];
  const float* Wq = (const float*)d_in[2];
  const float* Wv = (const float*)d_in[3];
  float* out = (float*)d_out;

  const int B = in_sizes[0] / (T_SZ * C_SZ);  // 512
  dim3 grid(B), block(NTHREADS);
  hipLaunchKernelGGL(fused_attn_head, grid, block, 0, stream, x, Wk, Wq, Wv, out);
}
